// StyleBlock2_65335042507468
// MI455X (gfx1250) — matrix operand order repaired and checked by reading, not compiled, not run on hardware
//
#include <hip/hip_runtime.h>
#include <hip/hip_bf16.h>

// Problem constants (match reference)
#define BB   16
#define LAT  512
#define CIN  512
#define WW   4096
#define KC   3
#define FF   512

// Tiling
#define MT 128   // f-tile
#define NT 128   // w-tile
#define KT 32    // c-chunk (one WMMA K)

typedef __attribute__((ext_vector_type(16))) __bf16        v16bf;
typedef __attribute__((ext_vector_type(8)))  float         v8f;
typedef __attribute__((ext_vector_type(4)))  unsigned int  v4u;
typedef __attribute__((ext_vector_type(8)))  int           i32x8;
typedef __attribute__((ext_vector_type(4)))  int           i32x4;

union Frag { v16bf bf; v4u u[2]; };

static __device__ __forceinline__ unsigned short f2bf(float f) {
    unsigned int u = __float_as_uint(f);
    unsigned int r = 0x7FFFu + ((u >> 16) & 1u);   // round-to-nearest-even
    return (unsigned short)((u + r) >> 16);
}

// -------- prep 1: affine[b,c] = latent[b,:] @ affine_w[:,c] + affine_b[c]
__global__ __launch_bounds__(256) void affine_kernel(
    const float* __restrict__ latent, const float* __restrict__ aw,
    const float* __restrict__ ab, float* __restrict__ affine) {
    int idx = blockIdx.x * 256 + threadIdx.x;       // B*CIN = 8192
    int b = idx / CIN, c = idx % CIN;
    float s = ab[c];
    for (int t = 0; t < LAT; ++t)
        s += latent[b * LAT + t] * aw[t * CIN + c];
    affine[idx] = s;
}

// -------- prep 2: demod[b,f] = rsqrt( sum_{k,c} kw[k,c,f]^2 * affine[b,c]^2 + 1e-8 )
__global__ __launch_bounds__(256) void demod_kernel(
    const float* __restrict__ kw, const float* __restrict__ affine,
    float* __restrict__ demod) {
    int idx = blockIdx.x * 256 + threadIdx.x;       // B*F = 8192
    int b = idx / FF, f = idx % FF;
    float s = 0.f;
    for (int c = 0; c < CIN; ++c) {
        float a = affine[b * CIN + c];
        float ksum = 0.f;
        for (int k = 0; k < KC; ++k) {
            float w = kw[((size_t)k * CIN + c) * FF + f];
            ksum += w * w;
        }
        s += ksum * a * a;
    }
    demod[idx] = rsqrtf(s + 1e-8f);
}

// -------- prep 3: Akt[k][f][c] = bf16(kernel_w[k][c][f])  (transposed, bf16)
__global__ __launch_bounds__(256) void pack_kernel(
    const float* __restrict__ kw, unsigned short* __restrict__ Akt) {
    int idx = blockIdx.x * 256 + threadIdx.x;       // KC*CIN*FF, f fastest (coalesced read)
    int f = idx % FF;
    int t = idx / FF;
    int c = t % CIN;
    int k = t / CIN;
    Akt[((size_t)k * FF + f) * CIN + c] = f2bf(kw[idx]);
}

// -------- main: modulated conv as 3-tap shifted bf16 WMMA GEMM + demod/bias/leaky epilogue
__global__ __launch_bounds__(256) void conv_kernel(
    const float* __restrict__ content, const float* __restrict__ affine,
    const unsigned short* __restrict__ Akt, const float* __restrict__ demod,
    const float* __restrict__ kb, float* __restrict__ out) {

    __shared__ __align__(16) unsigned short Xt[(NT + 2) * KT];   // [n=0..129][c], bf16 modulated input
    __shared__ __align__(16) unsigned short At[KC * MT * KT];    // [k][m][c],   bf16 kernel

    const int wb = blockIdx.x * NT;   // w base of Y tile
    const int fb = blockIdx.y * MT;   // f base of Y tile
    const int b  = blockIdx.z;

    const int tid    = threadIdx.x;
    const int lane   = tid & 31;
    const int wid    = tid >> 5;
    const int lm     = lane & 15;
    const int lh     = lane >> 4;
    const int wave_m = wid >> 2;      // 0..1 -> 4 m-subtiles each
    const int wave_n = wid & 3;       // 0..3 -> 2 n-subtiles each

    const v8f zero = {0.f, 0.f, 0.f, 0.f, 0.f, 0.f, 0.f, 0.f};
    v8f acc[4][2];
    for (int im = 0; im < 4; ++im)
        for (int in = 0; in < 2; ++in)
            acc[im][in] = zero;

    // LDS byte address of At (low 32 bits of shared pointer = LDS offset)
    const unsigned at_lds = (unsigned)(size_t)(&At[0]);

    for (int c0 = 0; c0 < CIN; c0 += KT) {
        // ---- A tiles for all 3 taps via Tensor Data Mover: one 3-D descriptor,
        //      tile = KT(c) x MT(f) x KC(k) from Akt[k][fb..fb+127][c0..c0+31].
        //      Issued by wave 0 only; overlaps with the X-tile VALU fill below.
        if (wid == 0) {
            unsigned long long ga =
                (unsigned long long)(const void*)(Akt + (size_t)fb * CIN + c0);
            v4u g0;
            g0.x = 0x1u;                                           // count=1, user mode
            g0.y = at_lds;                                         // lds_addr
            g0.z = (unsigned)(ga & 0xFFFFFFFFu);                   // global_addr[31:0]
            g0.w = (unsigned)((ga >> 32) & 0x01FFFFFFu)            // global_addr[56:32]
                   | (2u << 30);                                   // type=2 (image)

            const unsigned dim0 = CIN, dim1 = FF;
            const unsigned t0 = KT, t1 = MT, t2 = KC;
            const unsigned long long s0 = CIN;                     // stride dim0 (elems)
            const unsigned long long s1 = (unsigned long long)FF * CIN;
            i32x8 g1;
            g1[0] = (int)(1u << 16);                               // data_size=1 -> 2 bytes
            g1[1] = (int)((dim0 & 0xFFFFu) << 16);                 // tensor_dim0[15:0]
            g1[2] = (int)(((dim0 >> 16) & 0xFFFFu) | ((dim1 & 0xFFFFu) << 16));
            g1[3] = (int)(((dim1 >> 16) & 0xFFFFu) | ((t0 & 0xFFFFu) << 16));
            g1[4] = (int)((t1 & 0xFFFFu) | ((t2 & 0xFFFFu) << 16));
            g1[5] = (int)(s0 & 0xFFFFFFFFull);                     // dim0_stride[31:0]
            g1[6] = (int)(((s0 >> 32) & 0xFFFFull) | ((s1 & 0xFFFFull) << 16));
            g1[7] = (int)((s1 >> 16) & 0xFFFFFFFFull);             // dim1_stride[47:16]

            i32x4 g2 = {KC, 0, 0, 0};                              // tensor_dim2=3, rest unused
            i32x4 g3 = {0, 0, 0, 0};
            i32x8 g4 = {0, 0, 0, 0, 0, 0, 0, 0};                   // 5th SGPR group: unused/NULL
            __builtin_amdgcn_tensor_load_to_lds(g0, g1, g2, g3, g4, 0);
        }

        // ---- fill X tile: modulate + convert to bf16, haloed by 1 on each side
        for (int idx = tid; idx < (NT + 2) * KT; idx += 256) {
            int n = idx % (NT + 2);              // n fastest -> coalesced global reads
            int c = idx / (NT + 2);
            int w = wb - 1 + n;
            float v = 0.f;
            if (w >= 0 && w < WW) {
                const float* src = content + ((size_t)(b * CIN + c0 + c)) * WW + w;
                v = *src * affine[b * CIN + c0 + c];
                if (c0 + KT < CIN)
                    __builtin_prefetch(src + (size_t)KT * WW, 0, 1);  // next c-chunk
            }
            Xt[n * KT + c] = f2bf(v);
        }

        __builtin_amdgcn_s_wait_tensorcnt(0);    // wave 0: TDM done (no-op for others)
        __syncthreads();                         // publish At + Xt to all waves

        for (int k = 0; k < KC; ++k) {
            // B fragments: lane holds column n, halves c = 16*lh .. 16*lh+15
            Frag bfr[2];
            for (int in = 0; in < 2; ++in) {
                int col = (wave_n * 2 + in) * 16 + lm;
                const v4u* p = (const v4u*)(Xt + (col + k) * KT + lh * 16);
                bfr[in].u[0] = p[0];
                bfr[in].u[1] = p[1];
            }
            for (int im = 0; im < 4; ++im) {
                // A fragment: lane holds row m, halves c = 16*lh .. 16*lh+15
                int row = k * MT + (wave_m * 4 + im) * 16 + lm;
                Frag afr;
                afr.u[0] = *(const v4u*)(At + row * KT + lh * 16);
                afr.u[1] = *(const v4u*)(At + row * KT + lh * 16 + 8);
                for (int in = 0; in < 2; ++in) {
                    acc[im][in] = __builtin_amdgcn_wmma_f32_16x16x32_bf16(
                        false, afr.bf, false, bfr[in].bf,
                        (short)0, acc[im][in], false, false);
                }
            }
        }
        __syncthreads();
    }

    // ---- epilogue: demod scale + bias + leaky ReLU, store
    for (int im = 0; im < 4; ++im) {
        for (int in = 0; in < 2; ++in) {
            int wcol = wb + (wave_n * 2 + in) * 16 + lm;
            for (int r = 0; r < 8; ++r) {
                int f = fb + (wave_m * 4 + im) * 16 + r + lh * 8;
                float y = acc[im][in][r] * demod[b * FF + f] + kb[f];
                y = (y >= 0.f) ? y : 0.2f * y;
                out[((size_t)(b * FF + f)) * WW + wcol] = y;
            }
        }
    }
}

extern "C" void kernel_launch(void* const* d_in, const int* in_sizes, int n_in,
                              void* d_out, int out_size, void* d_ws, size_t ws_size,
                              hipStream_t stream) {
    const float* latent   = (const float*)d_in[0];   // [B, LAT]
    const float* content  = (const float*)d_in[1];   // [B, CIN, W]
    const float* affine_w = (const float*)d_in[2];   // [LAT, CIN]
    const float* affine_b = (const float*)d_in[3];   // [1, CIN]
    const float* kernel_w = (const float*)d_in[4];   // [K, CIN, F]
    const float* kernel_b = (const float*)d_in[5];   // [1, F, 1]
    float* out = (float*)d_out;                      // [B, F, W]

    char* ws = (char*)d_ws;
    float*          affine = (float*)ws;                    // B*CIN floats (32 KB)
    float*          demod  = (float*)(ws + 32768);          // B*F floats   (32 KB)
    unsigned short* Akt    = (unsigned short*)(ws + 65536); // K*F*CIN bf16 (1.5 MB)

    affine_kernel<<<(BB * CIN) / 256, 256, 0, stream>>>(latent, affine_w, affine_b, affine);
    demod_kernel <<<(BB * FF)  / 256, 256, 0, stream>>>(kernel_w, affine, demod);
    pack_kernel  <<<(KC * CIN * FF) / 256, 256, 0, stream>>>(kernel_w, Akt);

    dim3 grid(WW / NT, FF / MT, BB);   // 32 x 4 x 16
    conv_kernel<<<grid, 256, 0, stream>>>(content, affine, Akt, demod, kernel_b, out);
}
